// RelAttention_37426345017672
// MI455X (gfx1250) — matrix operand order repaired and checked by reading, not compiled, not run on hardware
//
#include <hip/hip_runtime.h>

// Problem constants (from reference): B=4, S=2048, D=1024, H=16, DH=64
#define B_  4
#define S_  2048
#define D_  1024
#define H_  16
#define DH_ 64
#define M_  (B_ * S_)        // 8192 tokens
#define REL_ (2 * S_ - 1)    // 4095 relative offsets

typedef __attribute__((ext_vector_type(16))) __bf16 v16bf;
typedef __attribute__((ext_vector_type(8)))  float  v8f;

union AB32 {                  // 32-byte WMMA operand (16 bf16)
    uint4          u4[2];
    unsigned short s[16];
    v16bf          v;
};

__device__ __forceinline__ unsigned short f2bf(float f) {
    unsigned int u = __float_as_uint(f);
    u += 0x7FFFu + ((u >> 16) & 1u);     // round-to-nearest-even
    return (unsigned short)(u >> 16);
}

__device__ __forceinline__ v8f wmma_bf16(const AB32& a, const AB32& b, v8f c) {
    return __builtin_amdgcn_wmma_f32_16x16x32_bf16(false, a.v, false, b.v,
                                                   (short)0, c, false, false);
}

// ---------------------------------------------------------------- casts
__global__ void cast_bf16_kernel(const float* __restrict__ src,
                                 unsigned short* __restrict__ dst, int n) {
    int i = blockIdx.x * blockDim.x + threadIdx.x;
    if (i < n) dst[i] = f2bf(src[i]);
}

// W [D,D] row-major fp32  ->  Wt [N,K] row-major bf16 (Wt[n][k] = W[k][n])
__global__ void transpose_cast_kernel(const float* __restrict__ W,
                                      unsigned short* __restrict__ Wt) {
    __shared__ float tile[32][33];
    const int bx = blockIdx.x * 32, by = blockIdx.y * 32;
    const int tx = threadIdx.x, ty = threadIdx.y;            // block (32,8)
    #pragma unroll
    for (int i = 0; i < 32; i += 8)
        tile[ty + i][tx] = W[(size_t)(by + ty + i) * D_ + bx + tx];
    __syncthreads();
    #pragma unroll
    for (int i = 0; i < 32; i += 8)
        Wt[(size_t)(bx + ty + i) * D_ + by + tx] = f2bf(tile[tx][ty + i]);
}

// ---------------------------------------------------------------- WMMA GEMM
// C[8192,1024] = A[8192,1024] * Bt^T   (A row-major bf16, Bt = B^T row-major bf16)
// Each wave computes a 64x32 register tile (4x2 WMMA tiles): 8 WMMAs per 12 b128
// loads per K-step; A operands reused 2x, B operands reused 4x.
// MODE 0: store bf16 into [B,H,S,DH]    (Q, K)
// MODE 1: store bf16 into [B,H,DH,S]    (V transposed, for P*V B-operand)
// MODE 2: store fp32 into [M,N]         (final output projection)
template <int MODE>
__device__ __forceinline__ void store_tile(void* Cout, int m0, int n0,
                                           int g, int ln, const v8f& c) {
    const int nn = n0 + ln;
    if (MODE == 2) {
        float* out = (float*)Cout;
        #pragma unroll
        for (int r = 0; r < 8; r++) {
            const int mm = m0 + r + 8 * g;                   // C: M = r + 8g
            out[(size_t)mm * D_ + nn] = c[r];
        }
    } else {
        unsigned short* out = (unsigned short*)Cout;
        const int hh = nn >> 6, dd = nn & 63;
        #pragma unroll
        for (int r = 0; r < 8; r++) {
            const int mm = m0 + r + 8 * g;
            const int bb = mm >> 11, ss = mm & (S_ - 1);
            size_t idx = (MODE == 0)
                ? (((size_t)(bb * H_ + hh) * S_ + ss) * DH_ + dd)
                : (((size_t)(bb * H_ + hh) * DH_ + dd) * S_ + ss);
            out[idx] = f2bf(c[r]);
        }
    }
}

template <int MODE>
__global__ __launch_bounds__(128, 1)
void wmma_gemm_kernel(const unsigned short* __restrict__ A,
                      const unsigned short* __restrict__ Bt,
                      void* __restrict__ Cout) {
    const int lane = threadIdx.x & 31;
    const int wave = threadIdx.x >> 5;
    const int g    = (lane >> 4) & 1;
    const int ln   = lane & 15;
    const int n0   = blockIdx.x * 32;
    const int m0   = (blockIdx.y * 4 + wave) * 64;

    const unsigned short* arow[4];
    #pragma unroll
    for (int i = 0; i < 4; i++)
        arow[i] = A + (size_t)(m0 + 16 * i + ln) * D_;
    const unsigned short* brow[2];
    #pragma unroll
    for (int j = 0; j < 2; j++)
        brow[j] = Bt + (size_t)(n0 + 16 * j + ln) * D_;

    v8f c[4][2];
    #pragma unroll
    for (int i = 0; i < 4; i++)
        #pragma unroll
        for (int j = 0; j < 2; j++) c[i][j] = v8f{};

    for (int k0 = 0; k0 < D_; k0 += 32) {
        AB32 a[4], b[2];
        #pragma unroll
        for (int i = 0; i < 4; i++) {
            a[i].u4[0] = *(const uint4*)(arow[i] + k0 + g * 8);      // A: K = g*8..+7
            a[i].u4[1] = *(const uint4*)(arow[i] + k0 + 16 + g * 8); //    K = 16+g*8..+7
        }
        #pragma unroll
        for (int j = 0; j < 2; j++) {
            b[j].u4[0] = *(const uint4*)(brow[j] + k0 + g * 8);      // B: K = g*8..+7
            b[j].u4[1] = *(const uint4*)(brow[j] + k0 + 16 + g * 8); //    K = 16+g*8..+7
        }
        #pragma unroll
        for (int i = 0; i < 4; i++)
            #pragma unroll
            for (int j = 0; j < 2; j++)
                c[i][j] = wmma_bf16(a[i], b[j], c[i][j]);
    }
    #pragma unroll
    for (int i = 0; i < 4; i++)
        #pragma unroll
        for (int j = 0; j < 2; j++)
            store_tile<MODE>(Cout, m0 + 16 * i, n0 + 16 * j, g, ln, c[i][j]);
}

// ---------------------------------------------------------------- fused attention
// One wave = one 16-query tile of one (b,h). Flash-style loop over 32-key chunks.
// Per chunk: 4 WMMAs QK^T, 6 WMMAs q*rel (48-offset span), 4 WMMAs P*V.
__global__ __launch_bounds__(128, 1)
void relattn_kernel(const unsigned short* __restrict__ Qb,
                    const unsigned short* __restrict__ Kb,
                    const unsigned short* __restrict__ Vt,
                    const unsigned short* __restrict__ relb,
                    unsigned short* __restrict__ attnout) {
    __shared__ float plds[4 * 16 * 32];                      // P staging, per wave
    const int lane = threadIdx.x & 31;
    const int wave = threadIdx.x >> 5;
    const int g    = (lane >> 4) & 1;
    const int ln   = lane & 15;
    const int bh   = blockIdx.y;                             // b*H + h
    const int i0   = (blockIdx.x * 4 + wave) * 16;

    const unsigned short* Qh = Qb + (size_t)bh * S_ * DH_;
    const unsigned short* Kh = Kb + (size_t)bh * S_ * DH_;
    const unsigned short* Vh = Vt + (size_t)bh * DH_ * S_;

    // Q tile as two A-operands (dh 0..31 and 32..63), resident for the whole row
    AB32 aQ0, aQ1;
    {
        const unsigned short* qrow = Qh + (size_t)(i0 + ln) * DH_;
        aQ0.u4[0] = *(const uint4*)(qrow + g * 8);
        aQ0.u4[1] = *(const uint4*)(qrow + 16 + g * 8);
        aQ1.u4[0] = *(const uint4*)(qrow + 32 + g * 8);
        aQ1.u4[1] = *(const uint4*)(qrow + 48 + g * 8);
    }

    float mrun[8], lrun[8];
    v8f acc[4] = {v8f{}, v8f{}, v8f{}, v8f{}};
    #pragma unroll
    for (int r = 0; r < 8; r++) { mrun[r] = -3.0e38f; lrun[r] = 0.f; }

    const float scale = 0.125f;                              // 1/sqrt(DH)
    float* P = &plds[wave * 16 * 32];

    for (int j0 = 0; j0 < S_; j0 += 32) {
        // ---- content scores: two 16-key subtiles, K-dim = DH = 64 (2 WMMAs each)
        v8f sc[2];
        #pragma unroll
        for (int st = 0; st < 2; st++) {
            const unsigned short* krow = Kh + (size_t)(j0 + st * 16 + ln) * DH_;
            AB32 b0, b1;
            b0.u4[0] = *(const uint4*)(krow + g * 8);
            b0.u4[1] = *(const uint4*)(krow + 16 + g * 8);
            b1.u4[0] = *(const uint4*)(krow + 32 + g * 8);
            b1.u4[1] = *(const uint4*)(krow + 48 + g * 8);
            v8f c = {};
            c = wmma_bf16(aQ0, b0, c);
            c = wmma_bf16(aQ1, b1, c);
            sc[st] = c;
        }
        // ---- qr over a 48-offset span covering both subtiles (3 tiles x 2 WMMAs)
        v8f qr[3];
        const int r0 = j0 - i0 + (S_ - 1) - 15;
        #pragma unroll
        for (int t = 0; t < 3; t++) {
            int rr = r0 + t * 16 + ln;
            rr = rr < 0 ? 0 : (rr > 2 * S_ - 2 ? 2 * S_ - 2 : rr);
            const unsigned short* rrow = relb + (size_t)rr * DH_;
            AB32 b0, b1;
            b0.u4[0] = *(const uint4*)(rrow + g * 8);
            b0.u4[1] = *(const uint4*)(rrow + 16 + g * 8);
            b1.u4[0] = *(const uint4*)(rrow + 32 + g * 8);
            b1.u4[1] = *(const uint4*)(rrow + 48 + g * 8);
            v8f c = {};
            c = wmma_bf16(aQ0, b0, c);
            c = wmma_bf16(aQ1, b1, c);
            qr[t] = c;
        }
        // ---- gather rel bias along anti-diagonals: bias(ii,jj) = qr[ii][jj-ii+15]
        #pragma unroll
        for (int st = 0; st < 2; st++) {
            #pragma unroll
            for (int r = 0; r < 8; r++) {
                const int ii   = r + 8 * g;
                const int jj   = ln;
                const int srcl = (lane & 16) | ((jj - ii + 15) & 15);
                float v0 = __shfl(qr[st][r],     srcl, 32);
                float v1 = __shfl(qr[st + 1][r], srcl, 32);
                float bias = (jj <= ii) ? v0 : v1;
                sc[st][r] = (sc[st][r] + bias) * scale;
            }
        }
        // ---- online softmax over this 32-key chunk (row stats per M = r+8g)
        #pragma unroll
        for (int r = 0; r < 8; r++) {
            float t = fmaxf(sc[0][r], sc[1][r]);
            #pragma unroll
            for (int off = 8; off >= 1; off >>= 1)
                t = fmaxf(t, __shfl_xor(t, off, 16));
            float mn   = fmaxf(mrun[r], t);
            float corr = __expf(mrun[r] - mn);
            mrun[r] = mn;
            float p0 = __expf(sc[0][r] - mn);
            float p1 = __expf(sc[1][r] - mn);
            float rs = p0 + p1;
            #pragma unroll
            for (int off = 8; off >= 1; off >>= 1)
                rs += __shfl_xor(rs, off, 16);
            lrun[r] = lrun[r] * corr + rs;
            #pragma unroll
            for (int d = 0; d < 4; d++) acc[d][r] = acc[d][r] * corr;
            const int ii = r + 8 * g;
            P[ii * 32 + ln]      = p0;
            P[ii * 32 + 16 + ln] = p1;
        }
        __syncthreads();
        // ---- repack P (C-layout, fp32 in LDS) into WMMA A-layout bf16
        AB32 aP;
        {
            const float* prow = &P[ln * 32];
            #pragma unroll
            for (int u = 0; u < 8; u++) aP.s[u]     = f2bf(prow[g * 8 + u]);
            #pragma unroll
            for (int u = 0; u < 8; u++) aP.s[8 + u] = f2bf(prow[16 + g * 8 + u]);
        }
        __syncthreads();
        // ---- O += P x V  (V stored transposed: contiguous K-runs per dh column)
        #pragma unroll
        for (int d = 0; d < 4; d++) {
            const unsigned short* vcol = Vh + (size_t)(d * 16 + ln) * S_ + j0;
            AB32 b;
            b.u4[0] = *(const uint4*)(vcol + g * 8);
            b.u4[1] = *(const uint4*)(vcol + 16 + g * 8);
            acc[d] = wmma_bf16(aP, b, acc[d]);
        }
    }

    // ---- normalize and store to [B,S,H*DH] bf16 (A-matrix for output projection)
    const int bb = bh / H_, hh = bh % H_;
    #pragma unroll
    for (int r = 0; r < 8; r++) {
        const int ii  = r + 8 * g;
        const float inv = 1.f / lrun[r];
        const size_t row = ((size_t)bb * S_ + i0 + ii) * D_ + hh * DH_;
        #pragma unroll
        for (int d = 0; d < 4; d++)
            attnout[row + d * 16 + ln] = f2bf(acc[d][r] * inv);
    }
}

// ---------------------------------------------------------------- launch
extern "C" void kernel_launch(void* const* d_in, const int* in_sizes, int n_in,
                              void* d_out, int out_size, void* d_ws, size_t ws_size,
                              hipStream_t stream) {
    const float* x   = (const float*)d_in[0];
    const float* Wq  = (const float*)d_in[1];
    const float* Wk  = (const float*)d_in[2];
    const float* Wv  = (const float*)d_in[3];
    const float* Wo  = (const float*)d_in[4];
    const float* rel = (const float*)d_in[5];

    char* w = (char*)d_ws;
    auto alloc = [&](size_t bytes) -> char* {
        char* p = w; w += (bytes + 255) & ~(size_t)255; return p;
    };
    unsigned short* xb   = (unsigned short*)alloc((size_t)M_ * D_ * 2);
    unsigned short* Wqt  = (unsigned short*)alloc((size_t)D_ * D_ * 2);
    unsigned short* Wkt  = (unsigned short*)alloc((size_t)D_ * D_ * 2);
    unsigned short* Wvt  = (unsigned short*)alloc((size_t)D_ * D_ * 2);
    unsigned short* Wot  = (unsigned short*)alloc((size_t)D_ * D_ * 2);
    unsigned short* relb = (unsigned short*)alloc((size_t)REL_ * DH_ * 2);
    unsigned short* Qb   = (unsigned short*)alloc((size_t)M_ * D_ * 2);
    unsigned short* Kb   = (unsigned short*)alloc((size_t)M_ * D_ * 2);
    unsigned short* Vtr  = (unsigned short*)alloc((size_t)M_ * D_ * 2);
    unsigned short* att  = (unsigned short*)alloc((size_t)M_ * D_ * 2);

    {
        int n = M_ * D_;
        cast_bf16_kernel<<<(n + 255) / 256, 256, 0, stream>>>(x, xb, n);
        n = REL_ * DH_;
        cast_bf16_kernel<<<(n + 255) / 256, 256, 0, stream>>>(rel, relb, n);
    }
    {
        dim3 tb(32, 8), tg(D_ / 32, D_ / 32);
        transpose_cast_kernel<<<tg, tb, 0, stream>>>(Wq, Wqt);
        transpose_cast_kernel<<<tg, tb, 0, stream>>>(Wk, Wkt);
        transpose_cast_kernel<<<tg, tb, 0, stream>>>(Wv, Wvt);
        transpose_cast_kernel<<<tg, tb, 0, stream>>>(Wo, Wot);
    }
    dim3 gg(D_ / 32, M_ / 256);               // 32 n-groups x 32 blocks (4 waves, 64x32 tile each)
    wmma_gemm_kernel<0><<<gg, 128, 0, stream>>>(xb, Wqt, Qb);
    wmma_gemm_kernel<0><<<gg, 128, 0, stream>>>(xb, Wkt, Kb);
    wmma_gemm_kernel<1><<<gg, 128, 0, stream>>>(xb, Wvt, Vtr);

    dim3 ag(S_ / 64, B_ * H_);                // 32 x 64 blocks, 4 q-tiles per block
    relattn_kernel<<<ag, 128, 0, stream>>>(Qb, Kb, Vtr, relb, att);

    wmma_gemm_kernel<2><<<gg, 128, 0, stream>>>(att, Wot, (float*)d_out);
}
